// MovieRecommendationModel_54700703482094
// MI455X (gfx1250) — matrix operand order repaired and checked by reading, not compiled, not run on hardware
//
#include <hip/hip_runtime.h>
#include <hip/hip_bf16.h>

typedef __attribute__((ext_vector_type(16))) __bf16       v16bf;
typedef __attribute__((ext_vector_type(8)))  float        v8f;
typedef __attribute__((ext_vector_type(4)))  unsigned int u32x4;

#define WMMA_BF16(a, b, c) \
  __builtin_amdgcn_wmma_f32_16x16x32_bf16(false, (a), false, (b), (short)0, (c), false, false)

constexpr int N_NODES = 32768;
constexpr int N_EDGES = 1048576;
constexpr int H1D = 128;
constexpr int H3D = 32;
constexpr int NU  = 16384;   // num_users (deterministic per reference)

struct BfPair { u32x4 lo, hi; };

// A-fragment: elements 0..7 = p[0..7], 8..15 = p[16..23]  (two b128 loads)
__device__ inline v16bf ld_a_split(const __bf16* p) {
  BfPair u{ *(const u32x4*)p, *(const u32x4*)(p + 16) };
  return __builtin_bit_cast(v16bf, u);
}
// B-fragment: elements 0..7 = p[0..7], 8..15 = p[16..23]  (two b128 loads)
__device__ inline v16bf ld_b_contig(const __bf16* p) {
  BfPair u{ *(const u32x4*)p, *(const u32x4*)(p + 16) };
  return __builtin_bit_cast(v16bf, u);
}

// ---------------- norm / layer-1 scalar pipeline ----------------

__global__ __launch_bounds__(256) void k_init_deg(float* deg) {
  int v = blockIdx.x * 256 + threadIdx.x;
  deg[v] = 1.0f;                               // self-loop
}

__global__ __launch_bounds__(256) void k_deg_edges(const long long* col, float* deg) {
  int e = blockIdx.x * 256 + threadIdx.x;
  atomicAdd(&deg[(int)col[e]], 1.0f);
}

__global__ __launch_bounds__(256) void k_dinv_s(const float* x, float* dinv_io, float* s) {
  int v = blockIdx.x * 256 + threadIdx.x;
  float di = rsqrtf(dinv_io[v]);               // deg >= 1 always
  dinv_io[v] = di;
  s[v] = di * di * x[v];                       // self-loop contribution
}

__global__ __launch_bounds__(256) void k_s_edges(const long long* row, const long long* col,
                                                 const float* x, const float* dinv, float* s) {
  int e = blockIdx.x * 256 + threadIdx.x;
  int r = (int)row[e], c = (int)col[e];
  atomicAdd(&s[c], dinv[r] * dinv[c] * x[r]);
}

__global__ __launch_bounds__(256) void k_h1(const float* s, const float* W1, const float* b1,
                                            __bf16* h1) {
  int idx = blockIdx.x * 256 + threadIdx.x;
  int v = idx >> 7, j = idx & 127;
  h1[idx] = (__bf16)fmaxf(fmaf(s[v], W1[j], b1[j]), 0.0f);
}

// -------- weight prep: W2 (128x128) -> W2T bf16 [n][k]; Wl (128x32) -> WlT bf16 [n][k] ----

__global__ __launch_bounds__(256) void k_prep_w(const float* W2, const float* Wl,
                                                __bf16* W2T, __bf16* WlT) {
  int idx = blockIdx.x * 256 + threadIdx.x;    // 0 .. 20479
  if (idx < 16384) {
    int n = idx >> 7, k = idx & 127;
    W2T[idx] = (__bf16)W2[k * 128 + n];
  } else {
    int j = idx - 16384;                       // 0 .. 4095
    int n = j >> 7, k = j & 127;
    WlT[j] = (__bf16)Wl[k * H3D + n];
  }
}

// ---------------- GEMM1: m2 = h1 @ W2  (32768x128 @ 128x128, bf16 WMMA, f32 out) --------

__global__ __launch_bounds__(256) void k_gemm1(const __bf16* h1, const __bf16* W2T, float* m2) {
  int lane   = threadIdx.x & 31;
  int waveId = blockIdx.x * 8 + (threadIdx.x >> 5);
  int tM = waveId >> 3;                        // 2048 row tiles
  int tN = waveId & 7;                         // 8 col tiles
  int m    = tM * 16 + (lane & 15);
  int ncol = tN * 16 + (lane & 15);
  int kA = (lane < 16) ? 0 : 8;
  int kB = (lane < 16) ? 0 : 8;
  v8f acc = {};
#pragma unroll
  for (int kk = 0; kk < 128; kk += 32) {
    v16bf aF = ld_a_split(h1 + m * H1D + kk + kA);
    v16bf bF = ld_b_contig(W2T + ncol * 128 + kk + kB);
    acc = WMMA_BF16(aF, bF, acc);
  }
  int rowBase = tM * 16 + ((lane < 16) ? 0 : 8);
#pragma unroll
  for (int r = 0; r < 8; ++r) m2[(rowBase + r) * H1D + ncol] = acc[r];
}

// ---------------- 128-wide GCN aggregation ----------------

__global__ __launch_bounds__(256) void k_agg_init(const float* m2, const float* dinv, float* agg) {
  int idx = blockIdx.x * 256 + threadIdx.x;
  float di = dinv[idx >> 7];
  agg[idx] = di * di * m2[idx];                // self-loop term
}

__global__ __launch_bounds__(256) void k_agg_edges(const long long* row, const long long* col,
                                                   const float* dinv, const float* m2, float* agg) {
  int gid  = blockIdx.x * 256 + threadIdx.x;
  int e    = gid >> 5;
  int lane = gid & 31;
  int r = (int)row[e], c = (int)col[e];
  float w = dinv[r] * dinv[c];
  const float4 mv = *(const float4*)(m2 + r * H1D + lane * 4);
  float* dst = agg + c * H1D + lane * 4;
  atomicAdd(dst + 0, w * mv.x);
  atomicAdd(dst + 1, w * mv.y);
  atomicAdd(dst + 2, w * mv.z);
  atomicAdd(dst + 3, w * mv.w);
}

// ------ GEMM2 fused: h3 = bf16(relu(relu(agg+b2) @ Wl + bl))  (32768x128 @ 128x32) ------

__global__ __launch_bounds__(256) void k_gemm2(const float* agg, const float* b2,
                                               const __bf16* WlT, const float* bl, __bf16* h3) {
  int lane   = threadIdx.x & 31;
  int waveId = blockIdx.x * 8 + (threadIdx.x >> 5);
  int tM = waveId >> 1;                        // 2048 row tiles
  int tN = waveId & 1;                         // 2 col tiles
  int m    = tM * 16 + (lane & 15);
  int ncol = tN * 16 + (lane & 15);
  int kA = (lane < 16) ? 0 : 8;
  int kB = (lane < 16) ? 0 : 8;
  v8f acc = {};
#pragma unroll
  for (int kk = 0; kk < 128; kk += 32) {
    v16bf aF, bF;
    const float* pa  = agg + m * H1D + kk + kA;
    const float* pb2 = b2 + kk + kA;
#pragma unroll
    for (int i = 0; i < 8; ++i) {
      aF[i]     = (__bf16)fmaxf(pa[i]      + pb2[i],      0.0f);
      aF[8 + i] = (__bf16)fmaxf(pa[16 + i] + pb2[16 + i], 0.0f);
    }
    bF = ld_b_contig(WlT + ncol * 128 + kk + kB);
    acc = WMMA_BF16(aF, bF, acc);
  }
  float bias = bl[ncol];
  int rowBase = tM * 16 + ((lane < 16) ? 0 : 8);
#pragma unroll
  for (int r = 0; r < 8; ++r)
    h3[(rowBase + r) * H3D + ncol] = (__bf16)fmaxf(acc[r] + bias, 0.0f);
}

// ------ GEMM3: out = users @ items^T  (16384x32)x(16384x32)^T, 2x2 tiles/wave, NT stores ------

__global__ __launch_bounds__(256) void k_gemm3(const __bf16* h3, float* out) {
  int lane   = threadIdx.x & 31;
  int waveId = blockIdx.x * 8 + (threadIdx.x >> 5);
  int wy = waveId >> 9;                        // 512 x 512 wave tiles
  int wx = waveId & 511;
  int mBase = wy * 32, nBase = wx * 32;
  const __bf16* users = h3;
  const __bf16* items = h3 + (long)NU * H3D;
  int kA = (lane < 16) ? 0 : 8;
  int kB = (lane < 16) ? 0 : 8;

  v16bf aF[2], bF[2];
#pragma unroll
  for (int sM = 0; sM < 2; ++sM) {
    int m = mBase + sM * 16 + (lane & 15);
    aF[sM] = ld_a_split(users + m * H3D + kA);       // K = 32 in one fragment
  }
#pragma unroll
  for (int sN = 0; sN < 2; ++sN) {
    int n = nBase + sN * 16 + (lane & 15);
    bF[sN] = ld_b_contig(items + n * H3D + kB);      // B[k][n] = items[n][k]
  }

  int hi8 = (lane < 16) ? 0 : 8;
  int nLo = lane & 15;
#pragma unroll
  for (int sM = 0; sM < 2; ++sM) {
#pragma unroll
    for (int sN = 0; sN < 2; ++sN) {
      v8f acc = {};
      acc = WMMA_BF16(aF[sM], bF[sN], acc);
      long rowBase = (long)(mBase + sM * 16 + hi8);
#pragma unroll
      for (int r = 0; r < 8; ++r)
        __builtin_nontemporal_store(acc[r], out + (rowBase + r) * (long)NU + nBase + sN * 16 + nLo);
    }
  }
}

// ---------------- launch ----------------

extern "C" void kernel_launch(void* const* d_in, const int* in_sizes, int n_in,
                              void* d_out, int out_size, void* d_ws, size_t ws_size,
                              hipStream_t stream) {
  const float*     x   = (const float*)d_in[0];
  const long long* ei  = (const long long*)d_in[1];
  const long long* row = ei;
  const long long* col = ei + N_EDGES;
  const float* W1 = (const float*)d_in[3];
  const float* b1 = (const float*)d_in[4];
  const float* W2 = (const float*)d_in[5];
  const float* b2 = (const float*)d_in[6];
  const float* Wl = (const float*)d_in[7];
  const float* bl = (const float*)d_in[8];
  float* out = (float*)d_out;

  char* ws = (char*)d_ws;                      // ~42.3 MB used, all re-initialized per call
  float*  dinv = (float*)(ws + 0);             // 128 KB (deg, then dinv in place)
  float*  s    = (float*)(ws + 131072);        // 128 KB
  __bf16* h1   = (__bf16*)(ws + 262144);       // 8 MB
  float*  m2   = (float*)(ws + 8650752);       // 16 MB
  float*  agg  = (float*)(ws + 25427968);      // 16 MB
  __bf16* h3   = (__bf16*)(ws + 42205184);     // 2 MB
  __bf16* W2T  = (__bf16*)(ws + 44302336);     // 32 KB (bf16, [n][k])
  __bf16* WlT  = (__bf16*)(ws + 44335104);     // 8 KB  (bf16, [n][k])

  k_prep_w   <<<80, 256, 0, stream>>>(W2, Wl, W2T, WlT);
  k_init_deg <<<N_NODES / 256, 256, 0, stream>>>(dinv);
  k_deg_edges<<<N_EDGES / 256, 256, 0, stream>>>(col, dinv);
  k_dinv_s   <<<N_NODES / 256, 256, 0, stream>>>(x, dinv, s);
  k_s_edges  <<<N_EDGES / 256, 256, 0, stream>>>(row, col, x, dinv, s);
  k_h1       <<<(N_NODES * H1D) / 256, 256, 0, stream>>>(s, W1, b1, h1);
  k_gemm1    <<<2048, 256, 0, stream>>>(h1, W2T, m2);
  k_agg_init <<<(N_NODES * H1D) / 256, 256, 0, stream>>>(m2, dinv, agg);
  k_agg_edges<<<(N_EDGES * 32) / 256, 256, 0, stream>>>(row, col, dinv, m2, agg);
  k_gemm2    <<<512, 256, 0, stream>>>(agg, b2, WlT, bl, h3);
  k_gemm3    <<<32768, 256, 0, stream>>>(h3, out);
}
